// DCNv2_2954937499734
// MI455X (gfx1250) — matrix operand order repaired and checked by reading, not compiled, not run on hardware
//
#include <hip/hip_runtime.h>
#include <hip/hip_bf16.h>
#include <math.h>

// ---------------------------------------------------------------------------
// DCNv2 forward for MI455X (gfx1250).
//   - offset conv (27ch 3x3)  : WMMA bf16 GEMM, M padded 27->32
//   - bilinear sample + einsum: fused WMMA bf16 GEMM (M=256, K=2304, N=32768)
// All operands resident in 192MB L2; bf16 operands, f32 accumulate.
// ---------------------------------------------------------------------------

typedef __attribute__((ext_vector_type(8)))  float  v8f;
typedef __attribute__((ext_vector_type(16))) __bf16 v16bf;

#define BATCH 8
#define CIN   256
#define COUT  256
#define HH    64
#define WW    64
#define HW    4096      // 64*64
#define KK    9
#define NT    128       // pixels per workgroup tile
#define CCH   32        // channels per K-chunk (WMMA bf16 K)
#define NCH   8         // K-chunks per tap (256/32)

__device__ __forceinline__ unsigned short f32_to_bf16_bits(float v) {
    unsigned int u = __builtin_bit_cast(unsigned int, v);
    u += 0x7fffu + ((u >> 16) & 1u);          // round-to-nearest-even
    return (unsigned short)(u >> 16);
}

// ---------------------------------------------------------------------------
// Kernel 1a: pack main weight (Cout,Cin,3,3) fp32 -> bf16 A-fragment order.
// wpack[k(9)][cchunk(8)][mtile(16)][lane(32)][h(16)]
// A 16x32 bf16 fragment, contiguous k order as in the B tile: lane = m + 16*(k32/16),
//   h = k32%16.
// ---------------------------------------------------------------------------
__global__ void pack_w(const float* __restrict__ weight,
                       unsigned short* __restrict__ wpack) {
    int t = blockIdx.x * 256 + threadIdx.x;
    const int total = KK * NCH * 16 * 32 * 16;   // 589824
    if (t >= total) return;
    int h    =  t        & 15;
    int lane = (t >> 4)  & 31;
    int mt   = (t >> 9)  & 15;
    int cc   = (t >> 13) & 7;
    int k    =  t >> 16;           // 0..8
    int m     = lane & 15;
    int group = lane >> 4;
    int k32   = group * 16 + h;
    int oc    = mt * 16 + m;
    int c     = cc * CCH + k32;
    wpack[t] = f32_to_bf16_bits(weight[((size_t)oc * CIN + c) * KK + k]);
}

// ---------------------------------------------------------------------------
// Kernel 1b: pack offset weight (27,Cin,3,3) fp32 -> bf16, M padded to 32.
// opack[k(9)][cchunk(8)][mtile(2)][lane(32)][h(16)]
// ---------------------------------------------------------------------------
__global__ void pack_ow(const float* __restrict__ ow,
                        unsigned short* __restrict__ opack) {
    int t = blockIdx.x * 256 + threadIdx.x;
    const int total = KK * NCH * 2 * 32 * 16;    // 73728
    if (t >= total) return;
    int h    =  t        & 15;
    int lane = (t >> 4)  & 31;
    int mt   = (t >> 9)  & 1;
    int cc   = (t >> 10) & 7;
    int k    =  t >> 13;           // 0..8
    int m     = lane & 15;
    int group = lane >> 4;
    int k32   = group * 16 + h;
    int oc    = mt * 16 + m;       // 0..31 (27..31 are padding)
    int c     = cc * CCH + k32;
    float w = (oc < 27) ? ow[((size_t)oc * CIN + c) * KK + k] : 0.0f;
    opack[t] = f32_to_bf16_bits(w);
}

// ---------------------------------------------------------------------------
// Kernel 2: offset conv via WMMA. grid: 256 blocks, block: 128 (4 waves).
// Wave w: m-tile = w&1, n-subtiles (w>>1)*4 .. +3.  B tile = zero-padded
// shifted x (ordinary im2col, no deformation). Output routed to
// dy/dx/sigmoid(mask) planes laid out [b][k][pixel], bias fused.
// ---------------------------------------------------------------------------
__global__ __launch_bounds__(128) void offset_conv_wmma(
        const float* __restrict__ x,
        const unsigned short* __restrict__ opack,
        const float* __restrict__ ob,
        float* __restrict__ dyp,
        float* __restrict__ dxp,
        float* __restrict__ maskp) {

    __shared__ __align__(32) unsigned short fragB[8 * 32 * 16]; // 8 KB

    int tile  = blockIdx.x;
    int b     = tile >> 5;
    int pbase = (tile & 31) * NT;
    int tid   = threadIdx.x;           // 0..127
    int lane  = tid & 31;
    int wave  = tid >> 5;              // 0..3
    int mt    = wave & 1;
    int n0    = (wave >> 1) * 4;

    // this thread's fixed pixel (one pixel per thread, 32 channels each)
    int pix  = tid;                    // 0..127
    int p    = pbase + pix;
    int ho   = p >> 6, wo = p & 63;
    int nsub = pix >> 4;
    int flb  = pix & 15;               // fragment lane low bits

    v8f acc[4];
    #pragma unroll
    for (int n = 0; n < 4; ++n) acc[n] = (v8f)(0.0f);

    for (int k = 0; k < KK; ++k) {
        int ki = k / 3, kj = k % 3;
        int y  = ho - 1 + ki;
        int xx = wo - 1 + kj;
        bool valid = ((unsigned)y < (unsigned)HH) & ((unsigned)xx < (unsigned)WW);
        const float* xrow = x + (((size_t)b * CIN) << 12) + y * WW + xx;

        for (int cc = 0; cc < NCH; ++cc) {
            int cbase = cc * CCH;
            // fill B tile: 32 channels for this pixel, pairs packed to b32
            #pragma unroll
            for (int i = 0; i < 16; ++i) {
                int c0 = 2 * i;
                float v0 = valid ? xrow[((size_t)(cbase + c0))     << 12] : 0.0f;
                float v1 = valid ? xrow[((size_t)(cbase + c0 + 1)) << 12] : 0.0f;
                unsigned int pk = (unsigned int)f32_to_bf16_bits(v0)
                                | ((unsigned int)f32_to_bf16_bits(v1) << 16);
                int fl = ((c0 >> 4) << 4) | flb;
                *(unsigned int*)&fragB[(nsub * 32 + fl) * 16 + (c0 & 15)] = pk;
            }
            __syncthreads();

            v16bf af = *(const v16bf*)(opack +
                ((size_t)(((k * NCH + cc) * 2 + mt) * 32 + lane) << 4));
            #pragma unroll
            for (int n = 0; n < 4; ++n) {
                v16bf bfm = *(const v16bf*)(&fragB[((n0 + n) * 32 + lane) * 16]);
                acc[n] = __builtin_amdgcn_wmma_f32_16x16x32_bf16(
                    false, af, false, bfm, (short)0, acc[n], false, false);
            }
            __syncthreads();
        }
    }

    // store: oc = mt*16 + 8*(lane/16) + r ; keep oc<27, route per channel kind
    int m0 = (lane >> 4) << 3;
    int nn = lane & 15;
    #pragma unroll
    for (int n = 0; n < 4; ++n) {
        int pp = pbase + (n0 + n) * 16 + nn;
        #pragma unroll
        for (int r = 0; r < 8; ++r) {
            int oc = mt * 16 + m0 + r;
            if (oc < 27) {
                float v = acc[n][r] + ob[oc];
                if (oc < 18) {
                    int kk = oc >> 1;
                    float* dst = (oc & 1) ? dxp : dyp;
                    dst[((size_t)b * KK + kk) * HW + pp] = v;
                } else {
                    int kk = oc - 18;
                    maskp[((size_t)b * KK + kk) * HW + pp] =
                        1.0f / (1.0f + __expf(-v));
                }
            }
        }
    }
}

// ---------------------------------------------------------------------------
// Kernel 3: fused bilinear sampling + WMMA GEMM.
// grid: 256 workgroups (one per 128-pixel tile), block: 512 (16 waves).
// Each wave owns one 16-row Cout tile; 8 N-subtiles of 16 pixels.
// ---------------------------------------------------------------------------
__global__ __launch_bounds__(512) void dcn_main(
        const float* __restrict__ x,
        const unsigned short* __restrict__ wpack,
        const float* __restrict__ bias,
        const float* __restrict__ dyp,
        const float* __restrict__ dxp,
        const float* __restrict__ maskp,
        float* __restrict__ out) {

    __shared__ __align__(32) unsigned short fragB[8 * 32 * 16]; // 8 KB
    __shared__ float pws[4][NT];   // bilinear weight * mask (0 if invalid)
    __shared__ int   pis[4][NT];   // clamped gather index (y*64+x)

    int tile  = blockIdx.x;            // 0..255
    int b     = tile >> 5;             // 32 tiles per image
    int pbase = (tile & 31) * NT;
    int tid   = threadIdx.x;
    int lane  = tid & 31;
    int wave  = tid >> 5;              // 0..15 -> Cout m-tile

    // sampler role: fixed pixel, contiguous 8-channel run per chunk
    int pix  = tid & 127;
    int cgrp = tid >> 7;               // 0..3 -> channels cgrp*8 .. +7
    int nsub = pix >> 4;
    int flb  = pix & 15;

    v8f acc[8];
    #pragma unroll
    for (int n = 0; n < 8; ++n) acc[n] = (v8f)(0.0f);

    for (int k = 0; k < KK; ++k) {
        int ki = k / 3, kj = k % 3;
        // ---- per-pixel bilinear precompute (threads 0..127) ----
        if (tid < NT) {
            int p  = pbase + tid;
            int ho = p >> 6, wo = p & 63;
            size_t oidx = ((size_t)b * KK + k) * HW + p;
            float py = dyp[oidx] + (float)(ho - 1 + ki);
            float px = dxp[oidx] + (float)(wo - 1 + kj);
            float m  = maskp[oidx];
            float yf = floorf(py), xf = floorf(px);
            float wy1 = py - yf, wx1 = px - xf;
            float wy0 = 1.0f - wy1, wx0 = 1.0f - wx1;
            int y0 = (int)yf, x0 = (int)xf;
            float wyv[2] = {wy0, wy1};
            float wxv[2] = {wx0, wx1};
            #pragma unroll
            for (int cy = 0; cy < 2; ++cy) {
                #pragma unroll
                for (int cx = 0; cx < 2; ++cx) {
                    int yi = y0 + cy, xi = x0 + cx;
                    bool valid = (yi >= 0) & (yi < HH) & (xi >= 0) & (xi < WW);
                    int yc = yi < 0 ? 0 : (yi > HH - 1 ? HH - 1 : yi);
                    int xc = xi < 0 ? 0 : (xi > WW - 1 ? WW - 1 : xi);
                    int ci = cy * 2 + cx;
                    pws[ci][tid] = valid ? (wyv[cy] * wxv[cx] * m) : 0.0f;
                    pis[ci][tid] = yc * WW + xc;
                }
            }
        }
        __syncthreads();

        // hoist this pixel's bilinear data into registers for all 8 chunks
        float w0 = pws[0][pix], w1 = pws[1][pix];
        float w2 = pws[2][pix], w3 = pws[3][pix];
        int   i0 = pis[0][pix], i1 = pis[1][pix];
        int   i2 = pis[2][pix], i3 = pis[3][pix];

        for (int cc = 0; cc < NCH; ++cc) {
            int cbase = cc * CCH + cgrp * 8;
            // ---- sampling: 8 contiguous channels, pairs packed to b32 ----
            #pragma unroll
            for (int i = 0; i < 4; ++i) {
                int c0 = cgrp * 8 + 2 * i;
                const float* xb0 =
                    x + ((size_t)(b * CIN + cbase + 2 * i) << 12);
                const float* xb1 = xb0 + HW;
                float v0 = w0 * xb0[i0] + w1 * xb0[i1]
                         + w2 * xb0[i2] + w3 * xb0[i3];
                float v1 = w0 * xb1[i0] + w1 * xb1[i1]
                         + w2 * xb1[i2] + w3 * xb1[i3];
                unsigned int pk = (unsigned int)f32_to_bf16_bits(v0)
                                | ((unsigned int)f32_to_bf16_bits(v1) << 16);
                int fl = ((c0 >> 4) << 4) | flb;
                *(unsigned int*)&fragB[(nsub * 32 + fl) * 16 + (c0 & 15)] = pk;
            }
            __syncthreads();

            // ---- WMMA: all lanes uniform (EXEC all-1s) ----
            v16bf af = *(const v16bf*)(wpack +
                ((size_t)(((k * NCH + cc) * 16 + wave) * 32 + lane) << 4));
            #pragma unroll
            for (int n = 0; n < 8; ++n) {
                v16bf bfm = *(const v16bf*)(&fragB[(n * 32 + lane) * 16]);
                acc[n] = __builtin_amdgcn_wmma_f32_16x16x32_bf16(
                    false, af, false, bfm, (short)0, acc[n], false, false);
            }
            __syncthreads();
        }
    }

    // ---- store: C/D layout: VGPR r -> M = r + 8*(lane/16), N = lane%16 ----
    int m0 = (lane >> 4) << 3;
    int nn = lane & 15;
    #pragma unroll
    for (int n = 0; n < 8; ++n) {
        #pragma unroll
        for (int r = 0; r < 8; ++r) {
            int oc = wave * 16 + m0 + r;
            int p  = pbase + n * 16 + nn;
            out[((size_t)(b * COUT + oc) << 12) + p] = acc[n][r] + bias[oc];
        }
    }
}

// ---------------------------------------------------------------------------
extern "C" void kernel_launch(void* const* d_in, const int* in_sizes, int n_in,
                              void* d_out, int out_size, void* d_ws, size_t ws_size,
                              hipStream_t stream) {
    const float* x        = (const float*)d_in[0];  // 8*256*64*64
    const float* weight   = (const float*)d_in[1];  // 256*256*3*3
    const float* bias     = (const float*)d_in[2];  // 256
    const float* offset_w = (const float*)d_in[3];  // 27*256*3*3
    const float* offset_b = (const float*)d_in[4];  // 27
    float* out = (float*)d_out;                     // 8*256*64*64

    // workspace layout
    const size_t nplane = (size_t)BATCH * KK * HW;       // 294912 floats
    float* dyp   = (float*)d_ws;
    float* dxp   = dyp + nplane;
    float* maskp = dxp + nplane;
    unsigned short* wpack = (unsigned short*)(maskp + nplane);   // 32B aligned
    unsigned short* opack = wpack + (size_t)KK * NCH * 16 * 32 * 16;

    // 1) pack weights into bf16 A-fragment order
    {
        const int total = KK * NCH * 16 * 32 * 16;  // 589824
        pack_w<<<(total + 255) / 256, 256, 0, stream>>>(weight, wpack);
    }
    {
        const int total = KK * NCH * 2 * 32 * 16;   // 73728
        pack_ow<<<(total + 255) / 256, 256, 0, stream>>>(offset_w, opack);
    }
    // 2) offset conv (WMMA) -> dy/dx/sigmoid(mask) planes
    offset_conv_wmma<<<(BATCH * HW) / NT, 128, 0, stream>>>(
        x, opack, offset_b, dyp, dxp, maskp);
    // 3) fused sample + WMMA GEMM
    dcn_main<<<(BATCH * HW) / NT, 512, 0, stream>>>(
        x, wpack, bias, dyp, dxp, maskp, out);
}
